// AuxillaryNetwork_68865505624589
// MI455X (gfx1250) — matrix operand order repaired and checked by reading, not compiled, not run on hardware
//
#include <hip/hip_runtime.h>

// MI455X / gfx1250, wave32, WMMA + Tensor Data Mover.
//
// Grouped tiny-MLP inference: 16 channels x [scalar -> 256 -> 4x(256x256) -> 1|2]
// over 16384 rows (~137 GFLOP, matrix-core bound). Hidden layers run as bf16
// WMMA (f32 accumulate). Activations stay resident in LDS. Per-layer weight
// slabs are split into two K=128 chunks and streamed L2 -> LDS by the TDM
// (tensor_load_to_lds, TENSORcnt), double-buffered so DMA overlaps WMMA.
// Each wave computes a 64x64 output tile (4x4 WMMA tiles) so both A and B
// fragments are reused 4x -> 1.0 ds_load_b128 per WMMA (LDS no longer limits).

typedef __attribute__((ext_vector_type(16))) __bf16 v16bf;
typedef __attribute__((ext_vector_type(8)))  float  v8f;
typedef unsigned int __attribute__((ext_vector_type(4))) u32x4;
typedef int          __attribute__((ext_vector_type(8))) i32x8;
typedef int          __attribute__((ext_vector_type(4))) i32x4;

#define HDIM   256      // hidden dim
#define BTILE  128      // batch rows per workgroup
#define NLAYER 4        // hidden layers
#define ASTR   264      // act LDS row stride (halves): 132 dwords ≡ 4 (mod 64) -> conflict-free
#define KC     128      // K-chunk (halves) streamed per TDM descriptor
#define WCS    136      // weight-chunk LDS row stride (halves): 68 dwords ≡ 4 (mod 64)
#define WCH_H  (HDIM * WCS)   // halves per weight chunk buffer (34816)
#define ZW     24       // z row width = R + 2P
#define BATCH  16384

__device__ __forceinline__ unsigned short f2bf(float f) {
    unsigned u = __float_as_uint(f);
    u += 0x7FFFu + ((u >> 16) & 1u);          // round-to-nearest-even
    return (unsigned short)(u >> 16);
}
__device__ __forceinline__ float bf2f(unsigned short h) {
    return __uint_as_float(((unsigned)h) << 16);
}

// ---------------------------------------------------------------------------
// Kernel 1: cast hidden-layer weights (Wr_h, Wc_h) f32 -> bf16 into workspace.
// ws layout: [2(group)][L=4][C=8][O=256][I=256] halves, row-major.
// ---------------------------------------------------------------------------
__global__ void convert_wh_bf16(const float* __restrict__ wr,
                                const float* __restrict__ wc,
                                unsigned short* __restrict__ out, int n_per) {
    int i = blockIdx.x * blockDim.x + threadIdx.x;
    if (i < n_per) {
        out[i]         = f2bf(wr[i]);
        out[n_per + i] = f2bf(wc[i]);
    }
}

// ---------------------------------------------------------------------------
// TDM descriptor: stream one K=128 weight chunk (256 rows x 256 bytes, global
// row stride 512 B) into LDS, with HW padding of 16 B after every 256 B so
// the landed LDS layout is [256][WCS] halves (bank-conflict-free B-frags).
// ---------------------------------------------------------------------------
__device__ __forceinline__ void tdm_issue_chunk(const unsigned short* gsrc,
                                                unsigned lds_byte_off) {
    unsigned long long ga = (unsigned long long)(uintptr_t)gsrc;
    // group0: count=1 | lds_addr | global_addr(57b) | type=2
    u32x4 g0 = { 1u,
                 lds_byte_off,
                 (unsigned)ga,
                 (unsigned)(ga >> 32) | (2u << 30) };
    // group1: data_size=4B(2) | pad_enable | pad_interval=256B(5) | pad_amount=4dw(3)
    //         tensor_dim0=64 dw, tensor_dim1=256, tile=64x256, dim0_stride=128 dw
    i32x8 g1 = { (int)((2u << 16) | (1u << 20) | (5u << 22) | (3u << 25)),
                 (int)(64u  << 16),   // [63:48]  tensor_dim0 lo
                 (int)(256u << 16),   // [79:64]=0, [95:80] tensor_dim1 lo
                 (int)(64u  << 16),   // [111:96]=0, [127:112] tile_dim0
                 (int)256,            // [143:128] tile_dim1, [159:144] tile_dim2=0
                 (int)128,            // [191:160] tensor_dim0_stride lo
                 0, 0 };
    i32x4 z4 = { 0, 0, 0, 0 };
#if __clang_major__ >= 23
    i32x8 z8 = { 0, 0, 0, 0, 0, 0, 0, 0 };
    __builtin_amdgcn_tensor_load_to_lds(g0, g1, z4, z4, z8, 0);
#else
    __builtin_amdgcn_tensor_load_to_lds(g0, g1, z4, z4, 0);
#endif
}

// ---------------------------------------------------------------------------
// Kernel 2: full MLP chain for one (channel, 128-row batch tile) per block.
// 256 threads = 8 waves tiling the 128x256 layer output as 2(M) x 4(N) wave
// groups; each wave owns a 64x64 tile = 4x4 WMMA accumulators.
// ---------------------------------------------------------------------------
__global__ __launch_bounds__(256)
void grouped_mlp_wmma(const float* __restrict__ z,
                      const float* __restrict__ Wr_in, const float* __restrict__ br_in,
                      const float* __restrict__ br_h,
                      const float* __restrict__ Wr_out, const float* __restrict__ br_out,
                      const float* __restrict__ Wc_in, const float* __restrict__ bc_in,
                      const float* __restrict__ bc_h,
                      const float* __restrict__ Wc_out, const float* __restrict__ bc_out,
                      const unsigned short* __restrict__ wh_bf,
                      float* __restrict__ out)
{
    extern __shared__ char smem[];
    unsigned short* sm_act = (unsigned short*)smem;              // [BTILE][ASTR]
    unsigned short* sm_w   = sm_act + BTILE * ASTR;              // 2 x [HDIM][WCS]
    float*          sm_x   = (float*)(sm_w + 2 * WCH_H);         // [BTILE]

    const int tid  = threadIdx.x;
    const int lane = tid & 31;
    const int wv   = tid >> 5;       // wave 0..7
    const int mg   = wv & 1;         // M group: rows mg*64 .. +64
    const int ng   = wv >> 1;        // N group: cols ng*64 .. +64
    const int tile = blockIdx.x;
    const int ch   = blockIdx.y;     // 0..7 real, 8..15 complex
    const int isC  = (ch >= 8);
    const int cc   = isC ? (ch - 8) : ch;
    const int b0   = tile * BTILE;

    const unsigned short* whbase =
        wh_bf + (isC ? (NLAYER * 8 * HDIM * HDIM) : 0) + cc * (HDIM * HDIM);
    const float* bh_base = isC ? bc_h : br_h;
    const unsigned smw_lds = (unsigned)(uintptr_t)(void*)sm_w;   // LDS byte offset

    // ---- kick off the first weight chunk DMA immediately (wave 0) ---------
    if (wv == 0)
        tdm_issue_chunk(whbase, smw_lds);

    // ---- per-row scalar channel input -> LDS (overlaps chunk-0 DMA) -------
    if (tid < BTILE) {
        const int b = b0 + tid;
        float x;
        if (!isC) {
            x = z[b * ZW + cc];
        } else {
            float zr = z[b * ZW + 8 + 2 * cc];
            float zi = z[b * ZW + 8 + 2 * cc + 1];
            x = zr * zr + zi * zi;
        }
        sm_x[tid] = x;
    }
    __syncthreads();

    // ---- input layer: act[m][tid] = relu(x[m]*W_in[tid] + b_in[tid]) ------
    {
        const float* Win = isC ? Wc_in : Wr_in;
        const float* bin = isC ? bc_in : br_in;
        const float wcol = Win[cc * HDIM + tid];
        const float bcol = bin[cc * HDIM + tid];
        #pragma unroll 4
        for (int m = 0; m < BTILE; ++m) {
            float v = fmaxf(sm_x[m] * wcol + bcol, 0.0f);
            sm_act[m * ASTR + tid] = f2bf(v);
        }
    }

    // fragment lane decompositions (ISA 16-bit A 16x32 / B 32x16 layouts)
    const int am = lane & 15, asel = lane >> 4;   // A: row-in-tile, K-half select
    const int bn = lane & 15, bkh  = lane >> 4;   // B: col-in-tile, K-half select

    union frag { uint4 u[2]; v16bf v; };
    v8f acc[4][4];

    // ---- 8 chunks = 4 layers x 2 K-halves, TDM double-buffered ------------
    for (int g = 0; g < 2 * NLAYER; ++g) {
        const int l  = g >> 1;
        const int kh = g & 1;

        if (wv == 0) {
            if (g + 1 < 2 * NLAYER) {
                const int l2 = (g + 1) >> 1, kh2 = (g + 1) & 1;
                tdm_issue_chunk(whbase + (size_t)l2 * (8 * HDIM * HDIM) + kh2 * KC,
                                smw_lds + (unsigned)(((g + 1) & 1) * (WCH_H * 2)));
                __builtin_amdgcn_s_wait_tensorcnt(1);   // chunk g has landed
            } else {
                __builtin_amdgcn_s_wait_tensorcnt(0);
            }
        }
        __syncthreads();   // chunk g visible to all waves (and act ready on g==0)

        if (kh == 0) {
            #pragma unroll
            for (int mt = 0; mt < 4; ++mt)
                #pragma unroll
                for (int n = 0; n < 4; ++n) { v8f zv = {}; acc[mt][n] = zv; }
        }

        const unsigned short* wb = sm_w + (size_t)(g & 1) * WCH_H;

        // ---- GEMM: 4 k-steps; per step 8 frag loads feed 16 WMMAs ---------
        #pragma unroll
        for (int ki = 0; ki < 4; ++ki) {
            const int k0 = kh * KC + ki * 32;      // global K into activations
            frag Af[4], Bf[4];
            #pragma unroll
            for (int mt = 0; mt < 4; ++mt) {
                const unsigned short* a =
                    sm_act + (mg * 64 + mt * 16 + am) * ASTR + k0 + asel * 8;
                Af[mt].u[0] = *(const uint4*)(a);
                Af[mt].u[1] = *(const uint4*)(a + 16);
            }
            #pragma unroll
            for (int n = 0; n < 4; ++n) {
                const unsigned short* bp =
                    wb + (ng * 64 + n * 16 + bn) * WCS + ki * 32 + bkh * 8;
                Bf[n].u[0] = *(const uint4*)(bp);
                Bf[n].u[1] = *(const uint4*)(bp + 16);
            }
            #pragma unroll
            for (int mt = 0; mt < 4; ++mt)
                #pragma unroll
                for (int n = 0; n < 4; ++n)
                    acc[mt][n] = __builtin_amdgcn_wmma_f32_16x16x32_bf16(
                        false, Af[mt].v, false, Bf[n].v, (short)0,
                        acc[mt][n], false, false);
        }

        // ---- layer epilogue after second K-chunk: bias + ReLU -> bf16 act --
        if (kh == 1) {
            const float* bl = bh_base + (size_t)(l * 8 + cc) * HDIM;
            #pragma unroll
            for (int mt = 0; mt < 4; ++mt) {
                const int rbase = mg * 64 + mt * 16 + (lane >> 4) * 8;
                #pragma unroll
                for (int n = 0; n < 4; ++n) {
                    const int col = ng * 64 + n * 16 + (lane & 15);
                    const float bv = bl[col];
                    #pragma unroll
                    for (int r = 0; r < 8; ++r) {
                        float v = fmaxf(acc[mt][n][r] + bv, 0.0f);
                        sm_act[(rbase + r) * ASTR + col] = f2bf(v);
                    }
                }
            }
        }
        __syncthreads();   // reads of this chunk done before its buffer is refilled
    }

    // ---- output head(s): out_dim 1 (real) or 2 (complex), dot over 256 ----
    {
        const float* Wout = isC ? Wc_out : Wr_out;
        const float* bout = isC ? bc_out : br_out;
        const int odim  = isC ? 2 : 1;
        const int total = BTILE * odim;
        for (int idx = tid; idx < total; idx += 256) {
            const int m = idx / odim;
            const int o = idx % odim;
            const float* wrow = Wout + (size_t)(cc * odim + o) * HDIM;
            float s = bout[cc * odim + o];
            const unsigned short* arow = sm_act + m * ASTR;
            #pragma unroll 8
            for (int i = 0; i < HDIM; ++i)
                s += bf2f(arow[i]) * wrow[i];
            const int b = b0 + m;
            if (!isC) out[(size_t)b * 8 + cc] = s;                               // real_lambda
            else      out[(size_t)BATCH * 8 * (1 + o) + (size_t)b * 8 + cc] = s; // mu / omega
        }
    }
}

// ---------------------------------------------------------------------------
extern "C" void kernel_launch(void* const* d_in, const int* in_sizes, int n_in,
                              void* d_out, int out_size, void* d_ws, size_t ws_size,
                              hipStream_t stream) {
    const float* z      = (const float*)d_in[0];
    const float* Wr_in  = (const float*)d_in[1];
    const float* br_in  = (const float*)d_in[2];
    const float* Wr_h   = (const float*)d_in[3];
    const float* br_h   = (const float*)d_in[4];
    const float* Wr_out = (const float*)d_in[5];
    const float* br_out = (const float*)d_in[6];
    const float* Wc_in  = (const float*)d_in[7];
    const float* bc_in  = (const float*)d_in[8];
    const float* Wc_h   = (const float*)d_in[9];
    const float* bc_h   = (const float*)d_in[10];
    const float* Wc_out = (const float*)d_in[11];
    const float* bc_out = (const float*)d_in[12];

    unsigned short* whb = (unsigned short*)d_ws;      // 2*4*8*256*256 halves = 8 MB
    const int n_per = NLAYER * 8 * HDIM * HDIM;       // 2,097,152 per group

    convert_wh_bf16<<<(n_per + 255) / 256, 256, 0, stream>>>(Wr_h, Wc_h, whb, n_per);

    const size_t lds_bytes =
        (size_t)BTILE * ASTR * sizeof(unsigned short)        // activations
        + (size_t)2 * WCH_H * sizeof(unsigned short)         // double-buffered weight chunks
        + (size_t)BTILE * sizeof(float);                     // scalar inputs
    dim3 grid(BATCH / BTILE, 16);                            // 128 tiles x 16 channels
    grouped_mlp_wmma<<<grid, 256, lds_bytes, stream>>>(
        z, Wr_in, br_in, br_h, Wr_out, br_out,
        Wc_in, bc_in, bc_h, Wc_out, bc_out,
        whb, (float*)d_out);
}
